// JointNetwork_5600637354581
// MI455X (gfx1250) — matrix operand order repaired and checked by reading, not compiled, not run on hardware
//
#include <hip/hip_runtime.h>
#include <hip/hip_bf16.h>
#include <math.h>

// ---- problem constants (match reference) ----
constexpr int HID   = 512;
constexpr int VOCAB = 1024;
constexpr int Bb    = 4;
constexpr int T     = 256;
constexpr int Uu    = 128;
// LDS row stride for the j tile: 512 + 8 bf16 = 1040 bytes. 1040/4 = 260 dwords,
// 260 mod 64 = 4 -> consecutive rows shift 4 banks; a 16-row b128 fragment read
// covers all 64 banks exactly twice per wave32 -> conflict-free.
constexpr int LDS_STRIDE = HID + 8;

typedef __attribute__((ext_vector_type(16))) __bf16 v16bf;
typedef __attribute__((ext_vector_type(4)))  __bf16 bf16x4;
typedef __attribute__((ext_vector_type(8)))  float  v8f;
typedef __attribute__((ext_vector_type(4)))  float  f32x4;
typedef __attribute__((ext_vector_type(4)))  unsigned int u32x4;

union Frag { u32x4 q[2]; v16bf v; };

// ---------------------------------------------------------------------------
// Kernel 0: W_joint fp32 -> bf16 (1 MB once per launch; L2-resident after)
// ---------------------------------------------------------------------------
__global__ void wj_to_bf16_kernel(const float* __restrict__ W,
                                  __bf16* __restrict__ Wb) {
    int i = blockIdx.x * blockDim.x + threadIdx.x;   // indexes groups of 4
    f32x4 w = ((const f32x4*)W)[i];
    bf16x4 o;
    o.x = (__bf16)w.x; o.y = (__bf16)w.y; o.z = (__bf16)w.z; o.w = (__bf16)w.w;
    ((bf16x4*)Wb)[i] = o;
}

// ---------------------------------------------------------------------------
// Kernel 1: fp32 projection  Y[row,h] = X[row,:] . W[h,:] + bias[h]
// One block per row, 256 threads, each thread does h=tid and h=tid+256.
// Exact fp32 (these feed tanh; only 0.6% of total FLOPs).
// ---------------------------------------------------------------------------
__global__ void __launch_bounds__(256) proj_kernel(const float* __restrict__ X,
                                                   const float* __restrict__ W,
                                                   const float* __restrict__ bias,
                                                   float* __restrict__ Y) {
    __shared__ float xs[HID];
    const int row = blockIdx.x;
    const int tid = threadIdx.x;
    xs[tid]       = X[(size_t)row * HID + tid];
    xs[tid + 256] = X[(size_t)row * HID + tid + 256];
    __syncthreads();

    const f32x4* x4 = (const f32x4*)xs;
    const f32x4* w0 = (const f32x4*)(W + (size_t)tid * HID);
    const f32x4* w1 = (const f32x4*)(W + (size_t)(tid + 256) * HID);
    float acc0 = 0.f, acc1 = 0.f;
#pragma unroll 4
    for (int k = 0; k < HID / 4; ++k) {
        f32x4 a = x4[k];
        f32x4 p = w0[k];
        f32x4 q = w1[k];
        acc0 += a.x * p.x + a.y * p.y + a.z * p.z + a.w * p.w;
        acc1 += a.x * q.x + a.y * q.y + a.z * q.z + a.w * q.w;
    }
    Y[(size_t)row * HID + tid]       = acc0 + bias[tid];
    Y[(size_t)row * HID + tid + 256] = acc1 + bias[tid + 256];
}

// ---------------------------------------------------------------------------
// Kernel 2: fused joint. One WG per (b,t). Phase A builds j=tanh(e+d) as a
// bf16 [128 x 512] tile in LDS; Phase B does the [128 x 1024 x 512] GEMM with
// v_wmma_f32_16x16x32_bf16. A-fragments come from LDS *every* pass (memory
// barriers below stop the compiler from caching them in scratch); B-fragments
// stream from the L2-resident bf16 W_joint.
// ---------------------------------------------------------------------------
__global__ void __launch_bounds__(256) joint_kernel(
        const float*  __restrict__ E,     // [B*T, HID]  fp32
        const float*  __restrict__ D,     // [B*U, HID]  fp32
        const __bf16* __restrict__ Wb,    // [VOCAB, HID] bf16
        const float*  __restrict__ bj,    // [VOCAB]
        float*        __restrict__ out) { // [B,T,U,VOCAB]
    extern __shared__ __bf16 jlds[];      // [Uu][LDS_STRIDE]

    const int bt  = blockIdx.x;           // 0..B*T-1
    const int b   = bt >> 8;              // T = 256
    const int tid = threadIdx.x;

    const float* Erow = E + (size_t)bt * HID;
    const float* Dbat = D + (size_t)b * Uu * HID;

    // ---- Phase A: j tile ----
    // 128 u x 512 h = 16384 float4 groups / 256 threads = 64 iters.
#pragma unroll 1
    for (int it = 0; it < (Uu * HID / 4) / 256; ++it) {
        int idx = it * 256 + tid;
        int u   = idx >> 7;               // 128 float4 groups per row
        int h   = (idx & 127) << 2;
        f32x4 d4 = *(const f32x4*)(Dbat + (size_t)u * HID + h);
        f32x4 e4 = *(const f32x4*)(Erow + h);
        bf16x4 j;
        j.x = (__bf16)tanhf(e4.x + d4.x);
        j.y = (__bf16)tanhf(e4.y + d4.y);
        j.z = (__bf16)tanhf(e4.z + d4.z);
        j.w = (__bf16)tanhf(e4.w + d4.w);
        *(bf16x4*)(jlds + u * LDS_STRIDE + h) = j;
    }
    __syncthreads();

    // ---- Phase B: WMMA GEMM ----
    const int lane = tid & 31;
    const int wv   = tid >> 5;            // 8 waves
    const int lhi  = lane >> 4;           // 0/1 half-wave
    const int llo  = lane & 15;

#pragma unroll 1
    for (int p = 0; p < 8; ++p) {
        // Barrier: nothing from previous passes may be kept live across here.
        asm volatile("" ::: "memory");

        const int v0  = p * 128 + wv * 16;
        const float bjv = bj[v0 + llo];   // C-layout: lane's column is fixed
        v8f acc[8];
        v8f seed = {bjv, bjv, bjv, bjv, bjv, bjv, bjv, bjv};
#pragma unroll
        for (int ut = 0; ut < 8; ++ut) acc[ut] = seed;

#pragma unroll 1
        for (int kc = 0; kc < HID / 32; ++kc) {
            // Barrier: keep the live set per kc step at acc(64)+A(8)+B(8)
            // VGPRs; prevents hoisting all LDS fragment loads (which would
            // spill to scratch, as observed in round 1).
            asm volatile("" ::: "memory");

            const int k0 = kc * 32;
            // B fragment: B = W_joint^T, so W_joint rows load directly as B
            // columns. Lane llo -> column v0+llo; two 8-elem K chunks at
            // k0+lhi*8 and +16 (same order as A).
            Frag Bf;
            const __bf16* wb = Wb + (size_t)(v0 + llo) * HID + k0 + lhi * 8;
            Bf.q[0] = *(const u32x4*)(wb);
            Bf.q[1] = *(const u32x4*)(wb + 16);
#pragma unroll
            for (int ut = 0; ut < 8; ++ut) {
                // A fragment: row = u, two 8-elem K chunks at k0+lhi*8, +16.
                const __bf16* ja =
                    jlds + (ut * 16 + llo) * LDS_STRIDE + k0 + lhi * 8;
                Frag Af;
                Af.q[0] = *(const u32x4*)(ja);
                Af.q[1] = *(const u32x4*)(ja + 16);
                acc[ut] = __builtin_amdgcn_wmma_f32_16x16x32_bf16(
                    false, Af.v, false, Bf.v, (short)0, acc[ut], false, false);
            }
        }

        // Epilogue: C/D layout -> lane column v0+llo, element i is row i (+8
        // for upper half-wave). Lanes 0-15 write 64 contiguous bytes per row.
#pragma unroll
        for (int ut = 0; ut < 8; ++ut) {
#pragma unroll
            for (int i = 0; i < 8; ++i) {
                int u = ut * 16 + i + (lhi << 3);
                out[(((size_t)bt * Uu + u) << 10) + v0 + llo] = acc[ut][i];
            }
        }
    }
}

// ---------------------------------------------------------------------------
extern "C" void kernel_launch(void* const* d_in, const int* in_sizes, int n_in,
                              void* d_out, int out_size, void* d_ws, size_t ws_size,
                              hipStream_t stream) {
    const float* enc   = (const float*)d_in[0];
    const float* dec   = (const float*)d_in[1];
    const float* W_enc = (const float*)d_in[2];
    const float* b_enc = (const float*)d_in[3];
    const float* W_dec = (const float*)d_in[4];
    const float* b_dec = (const float*)d_in[5];
    const float* W_j   = (const float*)d_in[6];
    const float* b_j   = (const float*)d_in[7];
    float* out = (float*)d_out;

    // workspace: E [B*T,HID] f32 (2 MB) | D [B*U,HID] f32 (1 MB) | Wb bf16 (1 MB)
    float*  E  = (float*)d_ws;
    float*  Dp = E + (size_t)Bb * T * HID;
    __bf16* Wb = (__bf16*)(Dp + (size_t)Bb * Uu * HID);

    wj_to_bf16_kernel<<<dim3((VOCAB * HID / 4) / 256), dim3(256), 0, stream>>>(W_j, Wb);
    proj_kernel<<<dim3(Bb * T), dim3(256), 0, stream>>>(enc, W_enc, b_enc, E);
    proj_kernel<<<dim3(Bb * Uu), dim3(256), 0, stream>>>(dec, W_dec, b_dec, Dp);

    size_t shmem = (size_t)Uu * LDS_STRIDE * sizeof(__bf16); // 133,120 B (<320 KB WGP)
    joint_kernel<<<dim3(Bb * T), dim3(256), shmem, stream>>>(E, Dp, Wb, b_j, out);
}
